// RGCN_38723425141326
// MI455X (gfx1250) — matrix operand order repaired and checked by reading, not compiled, not run on hardware
//
#include <hip/hip_runtime.h>

// ---------------------------------------------------------------------------
// R-GCN forward for MI455X (gfx1250, wave32, WMMA, async LDS copies).
//   R=2, N=6144, H1=H2=256
//   out1 = sum_r adj[r] @ w1_r          (w1_r = bc1-folded bw1)       38.7 GF
//   y_r  = relu(out1) @ w2_r            (w2_r = bc2-folded bw2)        1.6 GF
//   out2 = sum_r adj[r] @ y_r           (associativity rewrite)       38.7 GF
//   final = column-sums of [out1 | out2]
// adj (302 MB f32) is streamed twice with GLOBAL_LOAD_ASYNC_TO_LDS_B128
// (double-buffered, overlapped with WMMA), converted f32->bf16 at fragment
// load, and fed to v_wmma_f32_16x16x32_bf16 with f32 accumulation.
// ---------------------------------------------------------------------------

typedef __attribute__((ext_vector_type(16))) __bf16 v16bf;
typedef __attribute__((ext_vector_type(4)))  float  v4f;
typedef __attribute__((ext_vector_type(8)))  float  v8f;

#define NN 6144
#define H1 256
#define H2 256
#define AROWS 32          // block row tile
#define KSTEP 64          // k per pipeline stage
#define ASTRIDE 68        // 64 + 4 pad (f32 elems): row stride 272B, conflict-free b128
#define NTILES_K (NN / KSTEP)           // 96 per relation
#define NTILES_T (2 * NTILES_K)         // 192 total pipeline stages

// ------------------------- async / wait helpers ----------------------------
static __device__ __forceinline__ void async_copy_b128(unsigned lds_off,
                                                       unsigned long long gaddr) {
    asm volatile("global_load_async_to_lds_b128 %0, %1, off"
                 :: "v"(lds_off), "v"(gaddr) : "memory");
}
static __device__ __forceinline__ void async_copy_b128_off128(unsigned lds_off,
                                                              unsigned long long gaddr) {
    // INST_OFFSET applies to BOTH the LDS and global address (ISA 10.7 async)
    asm volatile("global_load_async_to_lds_b128 %0, %1, off offset:128"
                 :: "v"(lds_off), "v"(gaddr) : "memory");
}
static __device__ __forceinline__ void wait_async_le2() {
#if __has_builtin(__builtin_amdgcn_s_wait_asynccnt)
    __builtin_amdgcn_s_wait_asynccnt(2);
#else
    asm volatile("s_wait_asynccnt 0x2" ::: "memory");
#endif
}
static __device__ __forceinline__ void wait_async_le0() {
#if __has_builtin(__builtin_amdgcn_s_wait_asynccnt)
    __builtin_amdgcn_s_wait_asynccnt(0);
#else
    asm volatile("s_wait_asynccnt 0x0" ::: "memory");
#endif
}

// --------------------------- prep: w1_r transposed, bf16 -------------------
// w1bT[(r*256 + d)*6144 + j] = bf16( bc1[r][0]*bw1[0][j][d] + bc1[r][1]*bw1[1][j][d] )
__global__ __launch_bounds__(256)
void prep_w1T_kernel(const float* __restrict__ bw1, const float* __restrict__ bc1,
                     unsigned short* __restrict__ w1bT_u) {
    __bf16* w1bT = reinterpret_cast<__bf16*>(w1bT_u);
    size_t tid = (size_t)blockIdx.x * 256 + threadIdx.x;   // 2*6144*256 total
    int d = (int)(tid & 255);
    int j = (int)((tid >> 8) % NN);
    int r = (int)(tid >> 8) / NN;
    float v = bc1[r * 2 + 0] * bw1[(size_t)j * H1 + d]
            + bc1[r * 2 + 1] * bw1[(size_t)(NN + j) * H1 + d];
    w1bT[((size_t)(r * H1 + d)) * NN + j] = (__bf16)v;
}

// --------------------------- prep: w2_r, f32 -------------------------------
__global__ __launch_bounds__(256)
void prep_w2_kernel(const float* __restrict__ bw2, const float* __restrict__ bc2,
                    float* __restrict__ w2r) {
    size_t tid = (size_t)blockIdx.x * 256 + threadIdx.x;   // 2*256*256 total
    int r = (int)(tid >> 16);
    size_t rest = tid & 65535;
    w2r[tid] = bc2[r * 2 + 0] * bw2[rest] + bc2[r * 2 + 1] * bw2[65536 + rest];
}

// --------------------------- big GEMM: C = sum_r adj[r] @ B_r --------------
// adj: f32 [2][6144][6144].  BT: bf16 [2][256][6144] (BT[r][n][k] = B_r[k][n]).
// C: f32 [6144][256].
// Block: 256 thr = 8 waves; block tile 32 rows x 256 cols.
// Wave (rg,cg): rows rg*16..+16, cols cg*64..+64 (4 WMMA tiles).
// A tiles (32 x 64 f32) are double-buffered in LDS via async copies.
__global__ __launch_bounds__(256)
void gemm_adj_kernel(const float* __restrict__ adj,
                     const unsigned short* __restrict__ BT_u,
                     float* __restrict__ C) {
    const __bf16* BT = reinterpret_cast<const __bf16*>(BT_u);
    __shared__ __align__(16) float Af[2 * AROWS * ASTRIDE];

    const int tid  = threadIdx.x;
    const int wave = tid >> 5;
    const int lane = tid & 31;
    const int rg   = wave >> 2;        // 0..1
    const int cg   = wave & 3;         // 0..3
    const int l16  = lane & 15;
    const int hi   = lane >> 4;
    const int blockRow = blockIdx.x * AROWS;

    // cooperative staging coords: thread -> (row 0..31, 16B chunk 0..7)
    const int srow = tid >> 3;
    const int skc  = (tid & 7) << 2;   // f32 element offset within row, 0..28

    // LDS byte offsets of this thread's staging slots (flat addr low 32 bits
    // are the LDS offset, ISA 10.2 aperture mapping)
    const unsigned ldsoff0 =
        (unsigned)(unsigned long long)(void*)&Af[srow * ASTRIDE + skc];
    const unsigned ldsoff1 = ldsoff0 + (unsigned)(AROWS * ASTRIDE * 4);
    // this thread's global source offset within a plane (element units)
    const size_t gelem = (size_t)(blockRow + srow) * NN + skc;

    v8f acc[4];
    v8f zero = {0.f, 0.f, 0.f, 0.f, 0.f, 0.f, 0.f, 0.f};
#pragma unroll
    for (int t = 0; t < 4; ++t) acc[t] = zero;

    // ---- issue async copy of pipeline stage `itx` into buffer `buf` ----
    auto issue = [&](int itx, int buf) {
        const int rr  = (itx >= NTILES_K) ? 1 : 0;
        const int kk0 = (itx - rr * NTILES_K) * KSTEP;
        unsigned long long g = (unsigned long long)(adj + (size_t)rr * NN * NN
                                                        + gelem + kk0);
        unsigned lo = buf ? ldsoff1 : ldsoff0;
        async_copy_b128(lo, g);            // f32 elems [skc .. skc+3]
        async_copy_b128_off128(lo, g);     // f32 elems [32+skc .. 32+skc+3]
    };

    issue(0, 0);
    for (int it = 0; it < NTILES_T; ++it) {
        const int buf = it & 1;
        if (it + 1 < NTILES_T) {
            issue(it + 1, buf ^ 1);        // overlap copy with this tile's math
            wait_async_le2();              // our 2 ops for tile `it` complete
        } else {
            wait_async_le0();
        }
        __syncthreads();                   // publish tile `it` across waves

        const int rr  = (it >= NTILES_K) ? 1 : 0;
        const int kk0 = (it - rr * NTILES_K) * KSTEP;
        const __bf16* bplane = BT + (size_t)rr * H1 * NN;
        const float*  abuf   = &Af[buf * AROWS * ASTRIDE];
        const int     mloc   = rg * 16 + l16;

#pragma unroll
        for (int ks = 0; ks < KSTEP; ks += 32) {
            // A fragment (16x32 bf16): lanes 0-15 hold K 0-15 of row m,
            // lanes 16-31 hold K 16-31 (contiguous K, matching B);
            // convert f32->bf16 here (co-executes with the WMMA pipe).
            const float* fp = abuf + mloc * ASTRIDE + ks + 16 * hi;
            v4f f0 = *reinterpret_cast<const v4f*>(fp);
            v4f f1 = *reinterpret_cast<const v4f*>(fp + 4);
            v4f f2 = *reinterpret_cast<const v4f*>(fp + 8);
            v4f f3 = *reinterpret_cast<const v4f*>(fp + 12);
            v16bf afrag;
#pragma unroll
            for (int i = 0; i < 4; ++i) {
                afrag[i]      = (__bf16)f0[i];
                afrag[4 + i]  = (__bf16)f1[i];
                afrag[8 + i]  = (__bf16)f2[i];
                afrag[12 + i] = (__bf16)f3[i];
            }
#pragma unroll
            for (int t = 0; t < 4; ++t) {
                // B fragment (32x16 bf16): lane = col n, elems = 16 contiguous K
                const int n = cg * 64 + t * 16 + l16;
                const __bf16* bp = bplane + (size_t)n * NN + kk0 + ks + 16 * hi;
                v16bf bfrag = *reinterpret_cast<const v16bf*>(bp);
                acc[t] = __builtin_amdgcn_wmma_f32_16x16x32_bf16(
                    false, afrag, false, bfrag, (short)0, acc[t], false, false);
            }
        }
        __syncthreads();                   // WAR: done reading before overwrite
    }

    // epilogue: C/D layout — lane = col, VGPR i = row (i + 8*hi)
#pragma unroll
    for (int t = 0; t < 4; ++t) {
        const int n = cg * 64 + t * 16 + l16;
#pragma unroll
        for (int i = 0; i < 8; ++i) {
            const int m = blockRow + rg * 16 + hi * 8 + i;
            C[(size_t)m * H1 + n] = acc[t][i];
        }
    }
}

// --------------------------- y_r = relu(out1) @ w2_r, stored transposed ----
__global__ __launch_bounds__(256)
void relu_xw2_kernel(const float* __restrict__ out1, const float* __restrict__ w2r,
                     unsigned short* __restrict__ yT_u) {
    __bf16* yT = reinterpret_cast<__bf16*>(yT_u);
    __shared__ float xs[8 * H1];
    const int i0 = blockIdx.x * 8;
    for (int e = threadIdx.x; e < 8 * H1; e += 256) {
        float v = out1[(size_t)i0 * H1 + e];
        xs[e] = v > 0.f ? v : 0.f;
    }
    __syncthreads();
    const int n = threadIdx.x;
    for (int r = 0; r < 2; ++r) {
        float acc[8] = {0.f, 0.f, 0.f, 0.f, 0.f, 0.f, 0.f, 0.f};
        const float* w = w2r + (size_t)r * H1 * H2 + n;   // coalesced across threads
        for (int m = 0; m < H1; ++m) {
            float wv = w[(size_t)m * H2];
#pragma unroll
            for (int q = 0; q < 8; ++q) acc[q] += xs[q * H1 + m] * wv;
        }
#pragma unroll
        for (int q = 0; q < 8; ++q)
            yT[((size_t)(r * H2 + n)) * NN + i0 + q] = (__bf16)acc[q];
    }
}

// --------------------------- final_rep: column sums ------------------------
__global__ __launch_bounds__(256)
void colsum_kernel(const float* __restrict__ out1, const float* __restrict__ out2,
                   float* __restrict__ finalv) {
    const int d = blockIdx.x * 256 + threadIdx.x;   // 0..511
    const float* src = (d < H1) ? out1 : out2;
    const int col = d & (H1 - 1);
    float s = 0.f;
    for (int i = 0; i < NN; ++i) s += src[(size_t)i * H1 + col];
    finalv[d] = s;
}

// ---------------------------------------------------------------------------
extern "C" void kernel_launch(void* const* d_in, const int* in_sizes, int n_in,
                              void* d_out, int out_size, void* d_ws, size_t ws_size,
                              hipStream_t stream) {
    const float* adj = (const float*)d_in[0];   // (2, 6144, 6144)
    const float* bw1 = (const float*)d_in[1];   // (2, 6144, 256)
    const float* bc1 = (const float*)d_in[2];   // (2, 2)
    const float* bw2 = (const float*)d_in[3];   // (2, 256, 256)
    const float* bc2 = (const float*)d_in[4];   // (2, 2)

    // workspace carve-up (total ~19.4 MB)
    char* ws = (char*)d_ws;
    const size_t SZ_PLANE = (size_t)2 * 256 * NN * 2;   // 6,291,456 B (bf16 [2][256][6144])
    unsigned short* w1bT = (unsigned short*)ws;                       // layer-1 B^T bf16
    unsigned short* yT   = (unsigned short*)(ws + SZ_PLANE);          // layer-2 B^T bf16
    float*          out1 = (float*)(ws + 2 * SZ_PLANE);               // (6144,256) f32
    float*          w2r  = (float*)(ws + 2 * SZ_PLANE + (size_t)NN * H1 * 4); // (2,256,256) f32

    float* out2   = (float*)d_out;            // (6144, 256) — tuple elem 0
    float* finalv = out2 + (size_t)NN * H2;   // (1, 512)    — tuple elem 1

    prep_w1T_kernel<<<(2 * NN * H1) / 256, 256, 0, stream>>>(bw1, bc1, w1bT);
    prep_w2_kernel<<<(2 * H1 * H2) / 256, 256, 0, stream>>>(bw2, bc2, w2r);
    gemm_adj_kernel<<<NN / AROWS, 256, 0, stream>>>(adj, w1bT, out1);
    relu_xw2_kernel<<<NN / 8, 256, 0, stream>>>(out1, w2r, yT);
    gemm_adj_kernel<<<NN / AROWS, 256, 0, stream>>>(adj, yT, out2);
    colsum_kernel<<<2, 256, 0, stream>>>(out1, out2, finalv);
}
